// MultiHeadAttn_81011673137467
// MI455X (gfx1250) — matrix operand order repaired and checked by reading, not compiled, not run on hardware
//
#include <hip/hip_runtime.h>

// ---------------------------------------------------------------------------
// MI455X (gfx1250) multi-head attention: QKV GEMM + flash attention + out-proj
// + residual + LayerNorm. Matmuls via v_wmma_f32_16x16x32_bf16 (wave32);
// row-major tile staging via global_load_async_to_lds_b128 (ASYNCcnt).
// ---------------------------------------------------------------------------

typedef __attribute__((ext_vector_type(16))) __bf16 bf16x16;
typedef __attribute__((ext_vector_type(8)))  float  floatx8;

#define N_HEAD  16
#define D_HEAD  64
#define D_MODEL 1024
#define BATCH   2
#define SEQ     2048
#define MTOT    (BATCH * SEQ)   // 4096 rows

// Async copy of 16B chunks global->LDS. IOFFSET is added to BOTH the global
// and the LDS address (ISA 15.18.3 / 08_async_tensor §4.4), so one base pair
// covers consecutive chunks.
#define ASYNC_CP16_X2(lds32, gaddr64)                                          \
  asm volatile("global_load_async_to_lds_b128 %0, %1, off\n\t"                 \
               "global_load_async_to_lds_b128 %0, %1, off offset:16"           \
               :: "v"(lds32), "v"(gaddr64) : "memory")

#define ASYNC_CP16_X4(lds32, gaddr64)                                          \
  asm volatile("global_load_async_to_lds_b128 %0, %1, off\n\t"                 \
               "global_load_async_to_lds_b128 %0, %1, off offset:16\n\t"       \
               "global_load_async_to_lds_b128 %0, %1, off offset:32\n\t"       \
               "global_load_async_to_lds_b128 %0, %1, off offset:48"           \
               :: "v"(lds32), "v"(gaddr64) : "memory")

#define WAIT_ASYNC() asm volatile("s_wait_asynccnt 0x0" ::: "memory")

static __device__ __forceinline__ unsigned lds_addr32(const void* p) {
  return (unsigned)(unsigned long long)p;  // low 32 bits = LDS offset
}

// ---------------- WMMA fragment helpers (CDNA5 ISA §7.12.2 layouts) --------
// A 16x32 bf16: lane l holds row (l&15); VGPRs 0-3 hold K = base+0..7,
// VGPRs 4-7 hold K = base+16..23, where base = (l>=16) ? 8 : 0.
static __device__ __forceinline__ bf16x16
load_frag_a(const __bf16* base, int ld, int row0, int k0, int lane) {
  const __bf16* p = base + (size_t)(row0 + (lane & 15)) * ld + k0 + ((lane >> 4) << 3);
  bf16x16 a;
#pragma unroll
  for (int i = 0; i < 8; ++i) { a[i] = p[i]; a[8 + i] = p[16 + i]; }
  return a;
}

// B 32x16 bf16 from TRANSPOSED storage [N][K]: lane l holds column (l&15);
// elements 0-7 hold K = base+0..7, elements 8-15 hold K = base+16..23, base = (l>=16) ? 8 : 0.
static __device__ __forceinline__ bf16x16
load_frag_b(const __bf16* base, int ld, int n0, int k0, int lane) {
  const __bf16* p = base + (size_t)(n0 + (lane & 15)) * ld + k0 + ((lane >> 4) << 3);
  bf16x16 b;
#pragma unroll
  for (int i = 0; i < 8; ++i) { b[i] = p[i]; b[8 + i] = p[16 + i]; }
  return b;
}

static __device__ __forceinline__ floatx8
wmma_bf16(bf16x16 a, bf16x16 b, floatx8 c) {
  // (neg_a, A, neg_b, B, c_mod, C, reuse_a, reuse_b)
  return __builtin_amdgcn_wmma_f32_16x16x32_bf16(false, a, false, b, (short)0, c,
                                                 false, false);
}

// ---------------- fp32 -> bf16 conversion ----------------------------------
__global__ void cvt_bf16_kernel(const float* __restrict__ in,
                                __bf16* __restrict__ out, int n) {
  int i = blockIdx.x * blockDim.x + threadIdx.x;
  if (i < n) out[i] = (__bf16)in[i];
}

// ---------------- QKV projection GEMM: [4096,1024] x [1024,3072] -----------
// Block tile 128x128, K-step 32; 8 waves in 2x4, each wave 64x32 (8 frags).
__global__ __launch_bounds__(256) void qkv_gemm_kernel(
    const __bf16* __restrict__ A, const __bf16* __restrict__ Bw,
    const float* __restrict__ bias, __bf16* __restrict__ qb,
    __bf16* __restrict__ kb, __bf16* __restrict__ vb) {
  __shared__ __align__(16) __bf16 As[128][32];   // A tile row-major
  __shared__ __align__(16) __bf16 Bt[128][32];   // B tile transposed [N][K]
  const int tid = threadIdx.x, lane = tid & 31, wave = tid >> 5;
  const int mBase = blockIdx.y * 128, nBase = blockIdx.x * 128;
  const int mw = (wave >> 2) * 64, nw = (wave & 3) * 32;
  const int K = D_MODEL, Nld = 3 * N_HEAD * D_HEAD;

  floatx8 acc[4][2];
#pragma unroll
  for (int mi = 0; mi < 4; ++mi)
#pragma unroll
    for (int ni = 0; ni < 2; ++ni) acc[mi][ni] = (floatx8){};

  const int ar = tid >> 1, ac = (tid & 1) << 4;             // A-tile slot
  const unsigned lA = lds_addr32(&As[ar][ac]);
  for (int kBase = 0; kBase < K; kBase += 32) {
    __syncthreads();
    {  // A: 128 rows x 32 cols; async 2x16B per thread, LDS layout matches
      const __bf16* g = A + (size_t)(mBase + ar) * K + kBase + ac;
      __builtin_prefetch(g + 32, 0, 1);  // next K tile -> global_prefetch_b8
      ASYNC_CP16_X2(lA, (unsigned long long)g);
    }
    {  // B: 32 rows x 128 cols, transpose into Bt (VGPR path, ds_store)
      int kr = tid >> 3, nc = (tid & 7) << 4;
      const __bf16* g = Bw + (size_t)(kBase + kr) * Nld + nBase + nc;
#pragma unroll
      for (int i = 0; i < 16; ++i) Bt[nc + i][kr] = g[i];
    }
    WAIT_ASYNC();
    __syncthreads();
    bf16x16 af[4], bf[2];
#pragma unroll
    for (int mi = 0; mi < 4; ++mi) af[mi] = load_frag_a(&As[0][0], 32, mw + mi * 16, 0, lane);
#pragma unroll
    for (int ni = 0; ni < 2; ++ni) bf[ni] = load_frag_b(&Bt[0][0], 32, nw + ni * 16, 0, lane);
#pragma unroll
    for (int mi = 0; mi < 4; ++mi)
#pragma unroll
      for (int ni = 0; ni < 2; ++ni) acc[mi][ni] = wmma_bf16(af[mi], bf[ni], acc[mi][ni]);
  }

  // Epilogue: +bias, route to q/k/v bf16 buffers [4096, 1024]
  const int half = lane >> 4, nl = lane & 15;
#pragma unroll
  for (int mi = 0; mi < 4; ++mi)
#pragma unroll
    for (int ni = 0; ni < 2; ++ni) {
      int n = nBase + nw + ni * 16 + nl;
      int seg = n >> 10, col = n & 1023;
      __bf16* dst = (seg == 0) ? qb : (seg == 1 ? kb : vb);
      float bv = bias[n];
#pragma unroll
      for (int r = 0; r < 8; ++r) {
        int m = mBase + mw + mi * 16 + (half << 3) + r;
        dst[(size_t)m * D_MODEL + col] = (__bf16)(acc[mi][ni][r] + bv);
      }
    }
}

// ---------------- Flash attention -----------------------------------------
// grid = (T/128, B*H). WG = 256 threads (8 waves); wave w owns 16 query rows.
// Streams KV in 64-key tiles; online softmax; O accumulators in registers.
__global__ __launch_bounds__(256) void flash_attn_kernel(
    const __bf16* __restrict__ qb, const __bf16* __restrict__ kb,
    const __bf16* __restrict__ vb, const int* __restrict__ mask,
    __bf16* __restrict__ av) {
  __shared__ __align__(16) __bf16 Qs[128][64];
  __shared__ __align__(16) __bf16 Ks[64][64];   // [kv][d] == B-transposed for Q*K^T
  __shared__ __align__(16) __bf16 Vt[64][64];   // [d][kv] == B-transposed for P*V
  __shared__ __align__(16) __bf16 Ps[128][64];
  __shared__ float  biasS[64];

  const int tid = threadIdx.x, lane = tid & 31, wave = tid >> 5;
  const int bh = blockIdx.y;
  const int b = bh >> 4, h = bh & 15;   // q/k/v source (bh = b*H + h)
  const int ob = bh & 1;                // faithful: out/mask batch = bh % B
  const int oh = bh >> 1;               // faithful: out head = bh / B
  const int qBase = blockIdx.x * 128;
  const float scale = 0.125f;           // 1/sqrt(64)

  {  // Q tile: 128 rows x 64; async 4x16B per thread (row-major match)
    int r = tid >> 1, c = (tid & 1) << 5;
    const __bf16* g = qb + (size_t)(b * SEQ + qBase + r) * D_MODEL + h * D_HEAD + c;
    ASYNC_CP16_X4(lds_addr32(&Qs[r][c]), (unsigned long long)g);
  }

  const int mbw = wave * 16;
  floatx8 o[4];
#pragma unroll
  for (int j = 0; j < 4; ++j) o[j] = (floatx8){};
  float m_run[8], l_run[8];
#pragma unroll
  for (int r = 0; r < 8; ++r) { m_run[r] = -__builtin_inff(); l_run[r] = 0.f; }

  const int half = lane >> 4, nl = lane & 15;
  const int kr = tid >> 2, kc = (tid & 3) << 4;             // K/V tile slot
  const unsigned lK = lds_addr32(&Ks[kr][kc]);

  for (int kv = 0; kv < SEQ; kv += 64) {
    __syncthreads();  // all waves done reading Ks/Vt/Ps of previous step
    {  // K tile [kv][d]: async 2x16B per thread, LDS layout matches
      const __bf16* g = kb + (size_t)(b * SEQ + kv + kr) * D_MODEL + h * D_HEAD + kc;
      ASYNC_CP16_X2(lK, (unsigned long long)g);
    }
    {  // V tile transposed -> Vt[d][kv] (VGPR path, scattered ds_store)
      const __bf16* g = vb + (size_t)(b * SEQ + kv + kr) * D_MODEL + h * D_HEAD + kc;
#pragma unroll
      for (int i = 0; i < 16; ++i) Vt[kc + i][kr] = g[i];
    }
    if (tid < 64) {
      int mv = mask[ob * SEQ + kv + tid];
      biasS[tid] = mv ? 1.0f : -__builtin_inff();  // faithful: +1 kept, -inf masked
    }
    WAIT_ASYNC();     // covers Q tile (first iter) + this step's K tile
    __syncthreads();

    // ---- S = Q K^T (per-wave 16x64, 4 n-tiles, 2 K-steps over dh) ----
    bf16x16 qa0 = load_frag_a(&Qs[0][0], 64, mbw, 0, lane);
    bf16x16 qa1 = load_frag_a(&Qs[0][0], 64, mbw, 32, lane);
    floatx8 s[4];
#pragma unroll
    for (int j = 0; j < 4; ++j) {
      floatx8 cc = (floatx8){};
      cc = wmma_bf16(qa0, load_frag_b(&Ks[0][0], 64, j * 16, 0, lane), cc);
      cc = wmma_bf16(qa1, load_frag_b(&Ks[0][0], 64, j * 16, 32, lane), cc);
      s[j] = cc;
    }
    float bcol[4];
#pragma unroll
    for (int j = 0; j < 4; ++j) bcol[j] = biasS[j * 16 + nl];

    // ---- online softmax (C-layout: VGPR r <-> row mbw + half*8 + r) ----
#pragma unroll
    for (int r = 0; r < 8; ++r) {
      float mx = -__builtin_inff();
#pragma unroll
      for (int j = 0; j < 4; ++j) {
        s[j][r] = s[j][r] * scale + bcol[j];
        mx = fmaxf(mx, s[j][r]);
      }
#pragma unroll
      for (int off = 1; off < 16; off <<= 1) mx = fmaxf(mx, __shfl_xor(mx, off, 32));
      float mnew = fmaxf(m_run[r], mx);
      float d = m_run[r] - mnew;
      float alpha = (d == d) ? __expf(d) : 1.0f;   // NaN when both -inf
      m_run[r] = mnew;
      float acc = 0.f;
#pragma unroll
      for (int j = 0; j < 4; ++j) {
        float e = s[j][r] - mnew;
        float pv = (e == e) ? __expf(e) : 0.0f;
        s[j][r] = pv;
        acc += pv;
      }
#pragma unroll
      for (int off = 1; off < 16; off <<= 1) acc += __shfl_xor(acc, off, 32);
      l_run[r] = l_run[r] * alpha + acc;
#pragma unroll
      for (int j = 0; j < 4; ++j) o[j][r] *= alpha;  // rescale O accumulators
      // park P (bf16) in LDS for re-layout into A fragments
#pragma unroll
      for (int j = 0; j < 4; ++j)
        Ps[mbw + (half << 3) + r][j * 16 + nl] = (__bf16)s[j][r];
    }

    // wave-local LDS RAW: P stores -> P A-fragment loads (same wave only)
    asm volatile("s_wait_dscnt 0" ::: "memory");
    __builtin_amdgcn_wave_barrier();

    // ---- O += P V ----
    bf16x16 pa0 = load_frag_a(&Ps[0][0], 64, mbw, 0, lane);
    bf16x16 pa1 = load_frag_a(&Ps[0][0], 64, mbw, 32, lane);
#pragma unroll
    for (int j = 0; j < 4; ++j) {
      o[j] = wmma_bf16(pa0, load_frag_b(&Vt[0][0], 64, j * 16, 0, lane), o[j]);
      o[j] = wmma_bf16(pa1, load_frag_b(&Vt[0][0], 64, j * 16, 32, lane), o[j]);
    }
  }

  // ---- normalize, store attn_vec bf16 in faithful [ob, t, oh*64 + d] ----
#pragma unroll
  for (int r = 0; r < 8; ++r) {
    float inv = (l_run[r] > 0.f) ? 1.0f / l_run[r] : 0.0f;
    int t = qBase + mbw + (half << 3) + r;
#pragma unroll
    for (int j = 0; j < 4; ++j)
      av[(size_t)(ob * SEQ + t) * D_MODEL + oh * D_HEAD + j * 16 + nl] =
          (__bf16)(o[j][r] * inv);
  }
}

// ---------------- Output projection: [4096,1024] x [1024,1024] + residual --
__global__ __launch_bounds__(256) void oproj_gemm_kernel(
    const __bf16* __restrict__ A, const __bf16* __restrict__ Bw,
    const float* __restrict__ resid, float* __restrict__ y) {
  __shared__ __align__(16) __bf16 As[128][32];
  __shared__ __align__(16) __bf16 Bt[128][32];
  const int tid = threadIdx.x, lane = tid & 31, wave = tid >> 5;
  const int mBase = blockIdx.y * 128, nBase = blockIdx.x * 128;
  const int mw = (wave >> 2) * 64, nw = (wave & 3) * 32;
  const int K = D_MODEL, Nld = D_MODEL;

  floatx8 acc[4][2];
#pragma unroll
  for (int mi = 0; mi < 4; ++mi)
#pragma unroll
    for (int ni = 0; ni < 2; ++ni) acc[mi][ni] = (floatx8){};

  const int ar = tid >> 1, ac = (tid & 1) << 4;
  const unsigned lA = lds_addr32(&As[ar][ac]);
  for (int kBase = 0; kBase < K; kBase += 32) {
    __syncthreads();
    {
      const __bf16* g = A + (size_t)(mBase + ar) * K + kBase + ac;
      __builtin_prefetch(g + 32, 0, 1);
      ASYNC_CP16_X2(lA, (unsigned long long)g);
    }
    {
      int krr = tid >> 3, nc = (tid & 7) << 4;
      const __bf16* g = Bw + (size_t)(kBase + krr) * Nld + nBase + nc;
#pragma unroll
      for (int i = 0; i < 16; ++i) Bt[nc + i][krr] = g[i];
    }
    WAIT_ASYNC();
    __syncthreads();
    bf16x16 af[4], bf[2];
#pragma unroll
    for (int mi = 0; mi < 4; ++mi) af[mi] = load_frag_a(&As[0][0], 32, mw + mi * 16, 0, lane);
#pragma unroll
    for (int ni = 0; ni < 2; ++ni) bf[ni] = load_frag_b(&Bt[0][0], 32, nw + ni * 16, 0, lane);
#pragma unroll
    for (int mi = 0; mi < 4; ++mi)
#pragma unroll
      for (int ni = 0; ni < 2; ++ni) acc[mi][ni] = wmma_bf16(af[mi], bf[ni], acc[mi][ni]);
  }

  const int half = lane >> 4, nl = lane & 15;
#pragma unroll
  for (int mi = 0; mi < 4; ++mi)
#pragma unroll
    for (int ni = 0; ni < 2; ++ni) {
      int n = nBase + nw + ni * 16 + nl;
#pragma unroll
      for (int r = 0; r < 8; ++r) {
        int m = mBase + mw + mi * 16 + (half << 3) + r;
        size_t idx = (size_t)m * D_MODEL + n;
        y[idx] = acc[mi][ni][r] + resid[idx];
      }
    }
}

// ---------------- Row LayerNorm (eps = 1e-3) --------------------------------
__global__ __launch_bounds__(256) void ln_kernel(const float* __restrict__ y,
                                                 const float* __restrict__ gamma,
                                                 const float* __restrict__ beta,
                                                 float* __restrict__ out) {
  __shared__ float red[8];
  const int row = blockIdx.x, tid = threadIdx.x;
  const float* x = y + (size_t)row * D_MODEL;

  float s = 0.f;
  for (int i = tid; i < D_MODEL; i += 256) s += x[i];
#pragma unroll
  for (int off = 1; off < 32; off <<= 1) s += __shfl_xor(s, off, 32);
  if ((tid & 31) == 0) red[tid >> 5] = s;
  __syncthreads();
  float mu = 0.f;
#pragma unroll
  for (int i = 0; i < 8; ++i) mu += red[i];
  mu *= (1.0f / D_MODEL);
  __syncthreads();

  float v = 0.f;
  for (int i = tid; i < D_MODEL; i += 256) { float d = x[i] - mu; v += d * d; }
#pragma unroll
  for (int off = 1; off < 32; off <<= 1) v += __shfl_xor(v, off, 32);
  if ((tid & 31) == 0) red[tid >> 5] = v;
  __syncthreads();
  float var = 0.f;
#pragma unroll
  for (int i = 0; i < 8; ++i) var += red[i];
  var *= (1.0f / D_MODEL);
  float rstd = rsqrtf(var + 1e-3f);

  for (int i = tid; i < D_MODEL; i += 256)
    __builtin_nontemporal_store((x[i] - mu) * rstd * gamma[i] + beta[i],
                                &out[(size_t)row * D_MODEL + i]);
}

// ---------------- Launch ----------------------------------------------------
extern "C" void kernel_launch(void* const* d_in, const int* in_sizes, int n_in,
                              void* d_out, int out_size, void* d_ws, size_t ws_size,
                              hipStream_t stream) {
  const float* inp   = (const float*)d_in[0];
  const int*   amask = (const int*)d_in[1];
  const float* Wqkv  = (const float*)d_in[2];
  const float* bqkv  = (const float*)d_in[3];
  const float* Wo    = (const float*)d_in[4];
  const float* gamma = (const float*)d_in[5];
  const float* beta  = (const float*)d_in[6];
  float* out = (float*)d_out;

  char* ws = (char*)d_ws;
  __bf16* inp_bf  = (__bf16*)(ws + ((size_t)0  << 20));  // 8 MB
  __bf16* wqkv_bf = (__bf16*)(ws + ((size_t)8  << 20));  // 6 MB
  __bf16* wo_bf   = (__bf16*)(ws + ((size_t)14 << 20));  // 2 MB
  __bf16* qbuf    = (__bf16*)(ws + ((size_t)16 << 20));  // 8 MB
  __bf16* kbuf    = (__bf16*)(ws + ((size_t)24 << 20));  // 8 MB
  __bf16* vbuf    = (__bf16*)(ws + ((size_t)32 << 20));  // 8 MB
  __bf16* avbuf   = (__bf16*)(ws + ((size_t)40 << 20));  // 8 MB
  float*  ybuf    = (float*) (ws + ((size_t)48 << 20));  // 16 MB

  // fp32 -> bf16
  cvt_bf16_kernel<<<(MTOT * D_MODEL + 255) / 256, 256, 0, stream>>>(inp, inp_bf, MTOT * D_MODEL);
  cvt_bf16_kernel<<<(D_MODEL * 3 * N_HEAD * D_HEAD + 255) / 256, 256, 0, stream>>>(
      Wqkv, wqkv_bf, D_MODEL * 3 * N_HEAD * D_HEAD);
  cvt_bf16_kernel<<<(D_MODEL * D_MODEL + 255) / 256, 256, 0, stream>>>(Wo, wo_bf, D_MODEL * D_MODEL);

  // QKV projection: [4096,1024] x [1024,3072]
  qkv_gemm_kernel<<<dim3(3 * D_MODEL / 128, MTOT / 128), 256, 0, stream>>>(
      inp_bf, wqkv_bf, bqkv, qbuf, kbuf, vbuf);

  // Flash attention: 128 query rows per WG, per (b,h)
  flash_attn_kernel<<<dim3(SEQ / 128, BATCH * N_HEAD), 256, 0, stream>>>(
      qbuf, kbuf, vbuf, amask, avbuf);

  // Output projection + residual
  oproj_gemm_kernel<<<dim3(D_MODEL / 128, MTOT / 128), 256, 0, stream>>>(
      avbuf, wo_bf, inp, ybuf);

  // LayerNorm
  ln_kernel<<<MTOT, 256, 0, stream>>>(ybuf, gamma, beta, out);
}
